// NASCell_14834817040407
// MI455X (gfx1250) — compile-verified, matrix operand order checked by reading, not run on hardware
//
#include <hip/hip_runtime.h>

#define BB 8192
#define II 1024
#define HH 1024

typedef __attribute__((ext_vector_type(16))) __bf16 v16bf;
typedef __attribute__((ext_vector_type(8)))  __bf16 v8bf;
typedef __attribute__((ext_vector_type(4)))  __bf16 v4bf;
typedef __attribute__((ext_vector_type(2)))  __bf16 v2bf;
typedef __attribute__((ext_vector_type(8)))  float  v8f;
typedef __attribute__((ext_vector_type(4)))  float  v4f;
typedef __attribute__((ext_vector_type(2)))  float  v2f;

// Packed f32 -> bf16 conversions; lowers to v_cvt_pk_bf16_f32 on gfx1250 (RNE)
__device__ __forceinline__ unsigned int pk2bf(float a, float b) {
  v2f t = {a, b};
  return __builtin_bit_cast(unsigned int, __builtin_convertvector(t, v2bf));
}
__device__ __forceinline__ uint2 pk4bf(float4 v) {
  v4f t = {v.x, v.y, v.z, v.w};
  return __builtin_bit_cast(uint2, __builtin_convertvector(t, v4bf));
}

__device__ __forceinline__ float frcp(float x) { return __builtin_amdgcn_rcpf(x); }
__device__ __forceinline__ float sigf(float x) { return frcp(1.0f + __expf(-x)); }
__device__ __forceinline__ float tanhfast(float x) {
  // 1 - 2/(e^{2x}+1); saturates correctly at +-1 for large |x|
  return 1.0f - 2.0f * frcp(__expf(2.0f * x) + 1.0f);
}

// LDS row stride: 32 K-elements + 8 pad = 40 bf16 (80 B, 16B-aligned, bank-spread)
#define LDA 40
#define LDB 40

// sched_group_barrier masks
#define SG_WMMA     0x008
#define SG_VMEM_RD  0x020
#define SG_DS_READ  0x100

__global__ __launch_bounds__(256) void nascell_fused_wmma(
    const float* __restrict__ x, const float* __restrict__ m_prev,
    const float* __restrict__ c_prev, const float* __restrict__ w_m,
    const float* __restrict__ w_inputs, float* __restrict__ out_m,
    float* __restrict__ out_c) {
  __shared__ unsigned short lsA[128 * LDA];  // [row][k]  128x32 bf16 A tile
  __shared__ unsigned short lsB[128 * LDB];  // [col][k]  (8 gates x 16 cols) x 32 bf16

  const int tid  = threadIdx.x;
  const int lane = tid & 31;       // wave32
  const int wave = tid >> 5;       // 8 waves
  const int half = lane >> 4;      // half-wave selects K octet
  const int l15  = lane & 15;

  const int hBase   = blockIdx.x * 16;    // 64 h-tiles
  const int rowBase = blockIdx.y * 128;   // 64 row-tiles

  // stage-A mapping: 8 float4 across 32 K, 32 rows per pass, 4 passes
  const int atx = tid & 7;
  const int aty = tid >> 3;
  // stage-B mapping: 32 float4 across 128 cols, K-pairs, 2 passes
  const int btn = tid & 31;
  const int btk = tid >> 5;                               // 0..7 -> k-pair base
  const int bc  = btn * 4;                                // col 0..127 in tile
  const size_t bgcol = (size_t)(bc >> 4) * HH + hBase + (bc & 15);  // gate*H + h

  // acc[0..7]: gates (gate3 = m-phase part), acc[8]: gate3 x-phase part
  v8f acc[9];
  const v8f vzero = {0.f, 0.f, 0.f, 0.f, 0.f, 0.f, 0.f, 0.f};
#pragma unroll
  for (int i = 0; i < 9; ++i) acc[i] = vzero;

  // software-pipeline registers for the next K-step tile
  float4 ra[4];          // A: 4 rows-of-32 passes
  float4 rb0[2], rb1[2]; // B: 2 passes, rows k0 and k0+1

  auto load_tile = [&](const float* __restrict__ Ag, const float* __restrict__ Wg,
                       int kk) {
#pragma unroll
    for (int r = 0; r < 4; ++r)
      ra[r] = *(const float4*)(Ag + (size_t)(rowBase + aty + r * 32) * II + kk + atx * 4);
#pragma unroll
    for (int r = 0; r < 2; ++r) {
      const int k0 = btk * 2 + r * 16;
      rb0[r] = *(const float4*)(Wg + (size_t)(kk + k0) * (8 * HH) + bgcol);
      rb1[r] = *(const float4*)(Wg + (size_t)(kk + k0 + 1) * (8 * HH) + bgcol);
    }
  };

  auto store_tile = [&]() {
#pragma unroll
    for (int r = 0; r < 4; ++r)  // ds_store_b64, 4 bf16 along K
      *(uint2*)(&lsA[(aty + r * 32) * LDA + atx * 4]) = pk4bf(ra[r]);
#pragma unroll
    for (int r = 0; r < 2; ++r) {
      const int k0 = btk * 2 + r * 16;
      const float* lo = &rb0[r].x;
      const float* hi = &rb1[r].x;
#pragma unroll
      for (int j = 0; j < 4; ++j)  // k0 even -> 4B-aligned ds_store_b32
        *(unsigned int*)(&lsB[(bc + j) * LDB + k0]) = pk2bf(lo[j], hi[j]);
    }
  };

#pragma unroll  // fully unroll: p becomes a literal, gate-3 acc select folds away
  for (int p = 0; p < 2; ++p) {
    const float* __restrict__ Ag = p ? x : m_prev;        // [8192 x 1024]
    const float* __restrict__ Wg = p ? w_inputs : w_m;    // [1024 x 8192]
    load_tile(Ag, Wg, 0);
#pragma unroll 1
    for (int kk = 0; kk < HH; kk += 32) {
      store_tile();
      __syncthreads();
      if (kk + 32 < HH) load_tile(Ag, Wg, kk + 32);  // overlap with compute below

      // ---- A fragment: 16x32 bf16, lane=row&15, half selects K octets ----
      const int arow = wave * 16 + l15;
      const v8bf alo = *(const v8bf*)(&lsA[arow * LDA + half * 8]);
      const v8bf ahi = *(const v8bf*)(&lsA[arow * LDA + 16 + half * 8]);
      const v16bf afrag = __builtin_shufflevector(
          alo, ahi, 0, 1, 2, 3, 4, 5, 6, 7, 8, 9, 10, 11, 12, 13, 14, 15);

      // ---- B fragments + WMMAs ----
      v16bf bfr[8];
#pragma unroll
      for (int g = 0; g < 8; ++g) {
        const int col = g * 16 + l15;
        const v8bf blo = *(const v8bf*)(&lsB[col * LDB + half * 8]);
        const v8bf bhi = *(const v8bf*)(&lsB[col * LDB + 16 + half * 8]);
        bfr[g] = __builtin_shufflevector(
            blo, bhi, 0, 1, 2, 3, 4, 5, 6, 7, 8, 9, 10, 11, 12, 13, 14, 15);
      }
#pragma unroll
      for (int g = 0; g < 8; ++g) {
        if (g == 3 && p == 1)
          acc[8] = __builtin_amdgcn_wmma_f32_16x16x32_bf16(
              false, afrag, false, bfr[g], (short)0, acc[8], false, false);
        else
          acc[g] = __builtin_amdgcn_wmma_f32_16x16x32_bf16(
              false, afrag, false, bfr[g], (short)0, acc[g], false, false);
      }

      // ---- 3-deep fragment pipeline choreography ----
      // 1) next-tile global loads first (longest latency)
      __builtin_amdgcn_sched_group_barrier(SG_VMEM_RD, 8, 0);
      // 2) A frag (2) + B0,B1,B2 frags (6): 3 fragments in flight before wmma0
      __builtin_amdgcn_sched_group_barrier(SG_DS_READ, 8, 0);
      // 3) wmma g overlapped with load of frag g+3 -> partial dscnt waits
#pragma unroll
      for (int i = 0; i < 5; ++i) {
        __builtin_amdgcn_sched_group_barrier(SG_WMMA, 1, 0);
        __builtin_amdgcn_sched_group_barrier(SG_DS_READ, 2, 0);
      }
      // 4) drain: last three WMMAs
      __builtin_amdgcn_sched_group_barrier(SG_WMMA, 3, 0);

      __syncthreads();
    }
  }

  // ---- fused NAS-cell epilogue; C/D layout: VGPR v -> M=v+8*half, N=lane&15 ----
  const int col = hBase + l15;
#pragma unroll
  for (int v = 0; v < 8; ++v) {
    const int row = rowBase + wave * 16 + half * 8 + v;
    const size_t idx = (size_t)row * HH + col;
    const float cp = c_prev[idx];
    const float l1_0 = sigf(acc[0][v]);
    const float l1_1 = fmaxf(acc[1][v], 0.0f);
    const float l1_2 = sigf(acc[2][v]);
    const float l1_3 = fmaxf(acc[3][v] * acc[8][v], 0.0f);  // relu(ms3 * xs3)
    const float l1_4 = tanhfast(acc[4][v]);
    const float l1_5 = sigf(acc[5][v]);
    const float l1_6 = tanhfast(acc[6][v]);
    const float l1_7 = sigf(acc[7][v]);
    float l2_0       = tanhfast(l1_0 * l1_1);
    const float l2_1 = tanhfast(l1_2 + l1_3);
    const float l2_2 = tanhfast(l1_4 * l1_5);
    const float l2_3 = sigf(l1_6 + l1_7);
    l2_0             = tanhfast(l2_0 + cp);
    const float nc   = l2_0 * l2_1;
    const float l3_1 = tanhfast(l2_2 + l2_3);
    const float nm   = tanhfast(nc * l3_1);
    out_m[idx] = nm;
    out_c[idx] = nc;
  }
}

extern "C" void kernel_launch(void* const* d_in, const int* in_sizes, int n_in,
                              void* d_out, int out_size, void* d_ws, size_t ws_size,
                              hipStream_t stream) {
  const float* x        = (const float*)d_in[0];
  const float* m_prev   = (const float*)d_in[1];
  const float* c_prev   = (const float*)d_in[2];
  const float* w_m      = (const float*)d_in[3];
  const float* w_inputs = (const float*)d_in[4];
  float* out_m = (float*)d_out;                       // new_m [8192*1024]
  float* out_c = out_m + (size_t)BB * HH;             // new_c [8192*1024]
  dim3 grid(HH / 16, BB / 128);                       // 64 x 64 workgroups
  nascell_fused_wmma<<<grid, 256, 0, stream>>>(x, m_prev, c_prev, w_m, w_inputs,
                                               out_m, out_c);
}
